// GRU_RNN_16527034155632
// MI455X (gfx1250) — matrix operand order repaired and checked by reading, not compiled, not run on hardware
//
#include <hip/hip_runtime.h>
#include <hip/hip_fp16.h>

#define H      4096
#define SEQ    2048
#define HOR    64
#define TOTAL  (SEQ + HOR)
#define NWG    96
#define TPB    256
#define WPW    (TPB / 32)      // waves per workgroup = 8 -> 96*8 = 768 = 3*256 tiles
#define KBLK   (H / 32)        // 128 wmma k-steps per 16-row tile

typedef __attribute__((ext_vector_type(16))) _Float16 v16h;
typedef __attribute__((ext_vector_type(8)))  float    v8f;

union Frag {
    uint4 u[2];
    v16h  h;
};

// ---------------------------------------------------------------------------
// fp32 -> fp16 weight repack (runs every launch; ws is not preserved/trusted)
// ---------------------------------------------------------------------------
__global__ void convert_weights(const float* __restrict__ uWh,
                                const float* __restrict__ rWh,
                                const float* __restrict__ Wh,
                                unsigned short* __restrict__ out) {
    const size_t n      = (size_t)H * H;
    const size_t gidx   = (size_t)blockIdx.x * blockDim.x + threadIdx.x;
    const size_t stride = (size_t)gridDim.x * blockDim.x;
    _Float16* o = (_Float16*)out;
    for (size_t i = gidx; i < n; i += stride) o[i]         = (_Float16)uWh[i];
    for (size_t i = gidx; i < n; i += stride) o[n + i]     = (_Float16)rWh[i];
    for (size_t i = gidx; i < n; i += stride) o[2 * n + i] = (_Float16)Wh[i];
}

// ---------------------------------------------------------------------------
// zero h, y, and the grid-barrier words
// ---------------------------------------------------------------------------
__global__ void init_state(float* __restrict__ h_f32,
                           unsigned short* __restrict__ h_f16,
                           float* __restrict__ y_ws,
                           unsigned* __restrict__ cnt,
                           unsigned* __restrict__ gen) {
    const int i = blockIdx.x * blockDim.x + threadIdx.x;
    if (i < H) {
        h_f32[i] = 0.0f;
        h_f16[i] = 0;
    }
    if (i == 0) {
        *y_ws = 0.0f;
        *cnt  = 0u;
        *gen  = 0u;
    }
}

// ---------------------------------------------------------------------------
// grid-wide barrier: generation counter + device-scope atomics
// ---------------------------------------------------------------------------
__device__ __forceinline__ void gridBarrier(unsigned* cnt, unsigned* gen, unsigned nwg) {
    __syncthreads();
    if (threadIdx.x == 0) {
        __threadfence();
        volatile unsigned* vgen = (volatile unsigned*)gen;
        const unsigned g = *vgen;
        if (atomicAdd(cnt, 1u) == nwg - 1u) {
            atomicExch(cnt, 0u);
            __threadfence();
            atomicAdd(gen, 1u);
        } else {
            while (*vgen == g) { __builtin_amdgcn_s_sleep(2); }
            __threadfence();
        }
    }
    __syncthreads();
}

// ---------------------------------------------------------------------------
// persistent GRU kernel: 768 waves, one 16-row tile of one matrix per wave
// ---------------------------------------------------------------------------
__global__ void __launch_bounds__(TPB, 1)
gru_persist(const float* __restrict__ x_seq,
            const float* __restrict__ uWx, const float* __restrict__ ub,
            const float* __restrict__ rWx, const float* __restrict__ rb,
            const float* __restrict__ wx,  const float* __restrict__ bb,
            const float* __restrict__ V,   const float* __restrict__ c0,
            const unsigned short* __restrict__ Wf16,   // [3][H][H] row-major f16
            float* __restrict__ h_f32,
            unsigned short* __restrict__ h_f16,
            float* __restrict__ z,                      // [3][H]
            float* __restrict__ y_ws,
            unsigned* cnt, unsigned* gen,
            float* __restrict__ out) {
    __shared__ unsigned short sh_h[H];   // 8 KB: current hidden state, f16
    __shared__ float red[TPB];

    const int tid  = threadIdx.x;
    const int lane = tid & 31;
    const int gw   = blockIdx.x * WPW + (tid >> 5);   // 0..767
    const int mat  = gw >> 8;                         // 0..2
    const int tile = gw & 255;                        // 16-row tile
    const int r    = lane & 15;
    const int hi   = lane >> 4;

    // row of A this lane feeds (ISA 16-bit A layout: lanes 0-15 row M=lane
    // holding K=0..7,16..23; lanes 16-31 row M=lane-16 holding K=8..15,24..31)
    const _Float16* Arow =
        (const _Float16*)Wf16 + (size_t)mat * H * H + ((size_t)tile * 16 + r) * H;

    for (int step = 0; step < TOTAL; ++step) {
        // stage h (f16) into LDS: 8 KB, 16B per thread per pass
        for (int i = tid * 8; i < H; i += TPB * 8)
            *(uint4*)&sh_h[i] = *(const uint4*)&h_f16[i];
        __syncthreads();

        // z_tile[16] = W[tile rows, :] @ h   via 128 x wmma 16x16x32 f16
        v8f acc = {};
        #pragma unroll 4
        for (int kb = 0; kb < KBLK; ++kb) {
            const int k0 = kb * 32;
            __builtin_prefetch((const void*)(Arow + k0 + 512), 0, 3); // keep the L2 stream warm
            Frag a, b;
            a.u[0] = *(const uint4*)(Arow + k0 + hi * 8);        // K = k0+hi*8   .. +7
            a.u[1] = *(const uint4*)(Arow + k0 + 16 + hi * 8);   // K = k0+16+hi*8.. +7
            // B = h replicated to 16 cols: lanes 0-15 need h[k0..k0+7,k0+16..k0+23],
            // lanes 16-31 need h[k0+8..k0+15,k0+24..k0+31] (every column identical)
            b.u[0] = *(const uint4*)&sh_h[k0 + hi * 8];
            b.u[1] = *(const uint4*)&sh_h[k0 + 16 + hi * 8];
            acc = __builtin_amdgcn_wmma_f32_16x16x32_f16(
                false, a.h, false, b.h, (short)0, acc, false, false);
        }
        // every output column equals W@h; column N=0 lives in lanes 0 and 16
        if (r == 0) {
            float* zr = z + mat * H + tile * 16 + hi * 8;
            #pragma unroll
            for (int j = 0; j < 8; ++j) zr[j] = acc[j];
        }
        if (tid < WPW && lane == 0) __threadfence();
        gridBarrier(cnt, gen, NWG);

        // gates + state update + readout: block 0 only (4096 elems, 16/thread)
        if (blockIdx.x == 0) {
            const float x = (step < SEQ) ? x_seq[step] : *y_ws;
            float partial = 0.0f;
            for (int i = tid; i < H; i += TPB) {
                const float zu = z[i], zrr = z[H + i], zw = z[2 * H + i];
                const float u  = 1.0f / (1.0f + expf(-(uWx[i] * x + zu + ub[i])));
                const float rg = 1.0f / (1.0f + expf(-(rWx[i] * x + zrr + rb[i])));
                const float hh = tanhf(wx[i] * x + rg * zw + bb[i]);
                const float hn = (1.0f - u) * h_f32[i] + u * hh;
                h_f32[i] = hn;
                _Float16 hx = (_Float16)hn;
                h_f16[i] = *(unsigned short*)&hx;
                partial += V[i] * hn;
            }
            red[tid] = partial;
            __syncthreads();
            for (int s = TPB / 2; s > 0; s >>= 1) {
                if (tid < s) red[tid] += red[tid + s];
                __syncthreads();
            }
            if (tid == 0) {
                const float y = red[0] + c0[0];
                *y_ws = y;
                if (step >= SEQ) out[step - SEQ] = y;
            }
            __threadfence();
        }
        gridBarrier(cnt, gen, NWG);
    }
}

// ---------------------------------------------------------------------------
// host launch
// ---------------------------------------------------------------------------
extern "C" void kernel_launch(void* const* d_in, const int* in_sizes, int n_in,
                              void* d_out, int out_size, void* d_ws, size_t ws_size,
                              hipStream_t stream) {
    (void)in_sizes; (void)n_in; (void)out_size; (void)ws_size;
    const float* x_seq = (const float*)d_in[0];
    // d_in[1] = horizon (fixed 64 in reference; HOR hardcoded)
    const float* uWx = (const float*)d_in[2];
    const float* uWh = (const float*)d_in[3];
    const float* ub  = (const float*)d_in[4];
    const float* rWx = (const float*)d_in[5];
    const float* rWh = (const float*)d_in[6];
    const float* rb  = (const float*)d_in[7];
    const float* Wx  = (const float*)d_in[8];
    const float* Wh  = (const float*)d_in[9];
    const float* b   = (const float*)d_in[10];
    const float* V   = (const float*)d_in[11];
    const float* c   = (const float*)d_in[12];

    char* ws = (char*)d_ws;
    size_t off = 0;
    unsigned short* Wf16 = (unsigned short*)(ws + off); off += (size_t)3 * H * H * 2;
    float*          h32  = (float*)(ws + off);          off += (size_t)H * 4;
    unsigned short* h16  = (unsigned short*)(ws + off); off += (size_t)H * 2;
    float*          z    = (float*)(ws + off);          off += (size_t)3 * H * 4;
    float*          y_ws = (float*)(ws + off);          off += 4;
    unsigned*       cnt  = (unsigned*)(ws + off);       off += 4;
    unsigned*       gen  = (unsigned*)(ws + off);       off += 4;

    convert_weights<<<4096, TPB, 0, stream>>>(uWh, rWh, Wh, Wf16);
    init_state<<<(H + TPB - 1) / TPB, TPB, 0, stream>>>(h32, h16, y_ws, cnt, gen);
    gru_persist<<<NWG, TPB, 0, stream>>>(x_seq, uWx, ub, rWx, rb, Wx, b, V, c,
                                         Wf16, h32, h16, z, y_ws, cnt, gen,
                                         (float*)d_out);
}
